// MultiHeadAttention_52879637348925
// MI455X (gfx1250) — matrix operand order repaired and checked by reading, not compiled, not run on hardware
//
#include <hip/hip_runtime.h>
#include <hip/hip_bf16.h>

typedef _Float16 v16h __attribute__((ext_vector_type(16)));
typedef _Float16 v8h  __attribute__((ext_vector_type(8)));
typedef float    v8f  __attribute__((ext_vector_type(8)));

static constexpr int kDIM   = 1024;
static constexpr int kHEADS = 16;
static constexpr int kDK    = 64;
static constexpr int kB     = 2;
static constexpr int kLQ    = 2048;
static constexpr int kLK    = 2048;

union U16 { v16h v; v8h h2[2]; };

static __device__ __forceinline__ v8f wmma_f16(v16h a, v16h b, v8f c) {
    return __builtin_amdgcn_wmma_f32_16x16x32_f16(false, a, false, b, (short)0, c, false, false);
}

// ---------------------------------------------------------------------------
// Projection GEMM: Out = X @ W^T + bias, X:(4096 x 1024), W:(1024 x 1024)
// X_HALF: 0 = X is f32, 1 = X is f16
// OUT_MODE: 0 = f16 store to (b,h,l,d)   [Q,K]
//           1 = f16 store to (b,h,d,l)   [V transposed]
//           2 = f32 store row-major      [final output]
// Block: 256 threads (8 waves), tile 128x64, BK=32.
// Wave w owns row tile w (16 rows) x all 4 col tiles -> 4 wmma per k-step.
// ---------------------------------------------------------------------------
template<int X_HALF, int OUT_MODE>
__global__ __launch_bounds__(256) void proj_gemm(const void* __restrict__ Xv,
                                                 const float* __restrict__ W,
                                                 const float* __restrict__ bias,
                                                 void* __restrict__ Outv)
{
    __shared__ __align__(16) _Float16 lds_a[128 * 40];
    __shared__ __align__(16) _Float16 lds_b[64 * 40];

    const int tid  = threadIdx.x;
    const int lane = tid & 31;
    const int ln   = lane & 15;
    const int lg   = lane >> 4;
    const int w    = tid >> 5;          // row tile 0..7

    const int rowBase = blockIdx.x * 128;
    const int n0      = blockIdx.y * 64;

    const float*    Xf = (const float*)Xv;
    const _Float16* Xh = (const _Float16*)Xv;

    v8f acc[4] = {{}, {}, {}, {}};

    for (int k0 = 0; k0 < kDIM; k0 += 32) {
        // cooperative stage: A tile 128x32 (16/thread) and W tile 64x32 (8/thread)
        #pragma unroll
        for (int it = 0; it < 16; ++it) {
            int idx = tid + it * 256;
            int r = idx >> 5, c = idx & 31;
            float va = X_HALF ? (float)Xh[(size_t)(rowBase + r) * kDIM + k0 + c]
                              : Xf[(size_t)(rowBase + r) * kDIM + k0 + c];
            lds_a[r * 40 + c] = (_Float16)va;
        }
        #pragma unroll
        for (int it = 0; it < 8; ++it) {
            int idx = tid + it * 256;
            int r = idx >> 5, c = idx & 31;
            lds_b[r * 40 + c] = (_Float16)W[(size_t)(n0 + r) * kDIM + k0 + c];
        }
        __syncthreads();

        // A fragment (16x32), wave32 layout: lane ln = M row, k = lg*16 + g*8 + j
        U16 af;
        {
            const _Float16* ar = &lds_a[(w * 16 + ln) * 40 + lg * 16];
            af.h2[0] = *(const v8h*)(ar);
            af.h2[1] = *(const v8h*)(ar + 8);
        }
        // B fragments (32x16): lane ln = N col, k = lg*16 + h (16 contiguous)
        #pragma unroll
        for (int ct = 0; ct < 4; ++ct) {
            U16 bf;
            const _Float16* br = &lds_b[(ct * 16 + ln) * 40 + lg * 16];
            bf.h2[0] = *(const v8h*)(br);
            bf.h2[1] = *(const v8h*)(br + 8);
            acc[ct] = wmma_f16(af.v, bf.v, acc[ct]);
        }
        __syncthreads();
    }

    // epilogue
    const int mTile = rowBase + w * 16;
    #pragma unroll
    for (int ct = 0; ct < 4; ++ct) {
        int n = n0 + ct * 16 + ln;
        float bv = bias[n];
        #pragma unroll
        for (int r = 0; r < 8; ++r) {
            int m = mTile + lg * 8 + r;      // C layout: VGPR r -> row r / 8+r
            float val = acc[ct][r] + bv;
            if (OUT_MODE == 2) {
                ((float*)Outv)[(size_t)m * kDIM + n] = val;
            } else {
                int b = m >> 11, lpos = m & (kLQ - 1);   // Lseq = 2048
                int h = n >> 6,  d = n & 63;
                _Float16* Oo = (_Float16*)Outv;
                if (OUT_MODE == 0)
                    Oo[(((size_t)(b * kHEADS + h) * kLQ + lpos) * kDK) + d] = (_Float16)val;
                else
                    Oo[(((size_t)(b * kHEADS + h) * kDK + d) * kLK) + lpos] = (_Float16)val;
            }
        }
    }
}

// ---------------------------------------------------------------------------
// Flash attention: grid (LQ/64, B*H), 128 threads = 4 waves, wave owns 16 q rows.
// Qh,Kh: (b,h,l,d) f16; Vt: (b,h,d,l) f16; out Oh: (b, l, h*64+d) f16.
// ---------------------------------------------------------------------------
__global__ __launch_bounds__(128) void attn_kernel(const _Float16* __restrict__ Qh,
                                                   const _Float16* __restrict__ Kh,
                                                   const _Float16* __restrict__ Vt,
                                                   const int* __restrict__ am,
                                                   const int* __restrict__ kpm,
                                                   _Float16* __restrict__ Oh)
{
    __shared__ __align__(16) _Float16 pst[4][16 * 32];

    const int lane = threadIdx.x & 31;
    const int ln   = lane & 15;
    const int lg   = lane >> 4;
    const int w    = threadIdx.x >> 5;
    const int bh   = blockIdx.y;          // b*HEADS + h
    const int b    = bh >> 4;
    const int h    = bh & 15;
    const int q0   = blockIdx.x * 64 + w * 16;

    // resident Q A-fragments (16 rows x 64 dk = two 16x32 frags)
    U16 a0, a1;
    {
        const _Float16* qr = Qh + ((size_t)bh * kLQ + q0 + ln) * kDK + lg * 16;
        a0.h2[0] = *(const v8h*)(qr);
        a0.h2[1] = *(const v8h*)(qr + 8);
        a1.h2[0] = *(const v8h*)(qr + 32);
        a1.h2[1] = *(const v8h*)(qr + 40);
    }

    v8f o0 = {}, o1 = {}, o2 = {}, o3 = {};
    float mrow[8], lrow[8];
    #pragma unroll
    for (int r = 0; r < 8; ++r) { mrow[r] = -__builtin_inff(); lrow[r] = 0.0f; }

    for (int j = 0; j < kLK; j += 32) {
        const _Float16* kb = Kh + ((size_t)bh * kLK + j) * kDK;
        const _Float16* vb = Vt + (size_t)bh * kDK * kLK + j;

        // prefetch next chunk's K/V cachelines (global_prefetch_b8) to hide
        // L2 latency behind the softmax trans/VALU work below
        if (j + 32 < kLK) {
            __builtin_prefetch(kb + 32 * kDK + (size_t)lane * 64, 0, 0);
            __builtin_prefetch(vb + 32 + (size_t)(lane & 7) * 8 * kLK, 0, 0);
        }

        // ---- scores: two 16x16 C tiles over 32 keys, K(dk)=64 via 2 wmma each
        v8f s0 = {}, s1 = {};
        {
            v16h k00 = *(const v16h*)(kb + (size_t)ln * kDK + lg * 16);
            v16h k01 = *(const v16h*)(kb + (size_t)ln * kDK + 32 + lg * 16);
            s0 = wmma_f16(a0.v, k00, s0);
            s0 = wmma_f16(a1.v, k01, s0);
            v16h k10 = *(const v16h*)(kb + (size_t)(16 + ln) * kDK + lg * 16);
            v16h k11 = *(const v16h*)(kb + (size_t)(16 + ln) * kDK + 32 + lg * 16);
            s1 = wmma_f16(a0.v, k10, s1);
            s1 = wmma_f16(a1.v, k11, s1);
        }

        // ---- scale + masks, branchless: bitwise OR of unconditional loads
        const int key0 = j + ln;
        const int key1 = j + 16 + ln;
        const int kp0 = kpm[b * kLK + key0];
        const int kp1 = kpm[b * kLK + key1];
        float alpha[8];
        #pragma unroll
        for (int r = 0; r < 8; ++r) {
            int q = q0 + lg * 8 + r;
            int msk0 = kp0 | am[(size_t)q * kLK + key0];
            int msk1 = kp1 | am[(size_t)q * kLK + key1];
            s0[r] = (msk0 != 0) ? -1e30f : s0[r] * 0.125f;
            s1[r] = (msk1 != 0) ? -1e30f : s1[r] * 0.125f;
        }

        // ---- online softmax: row max / exp / row sum (xor-reduce in 16-lane half)
        #pragma unroll
        for (int r = 0; r < 8; ++r) {
            float t = fmaxf(s0[r], s1[r]);
            t = fmaxf(t, __shfl_xor(t, 1, 32));
            t = fmaxf(t, __shfl_xor(t, 2, 32));
            t = fmaxf(t, __shfl_xor(t, 4, 32));
            t = fmaxf(t, __shfl_xor(t, 8, 32));
            float mn = fmaxf(mrow[r], t);
            alpha[r] = __expf(mrow[r] - mn);
            mrow[r] = mn;
            s0[r] = __expf(s0[r] - mn);
            s1[r] = __expf(s1[r] - mn);
            float rs = s0[r] + s1[r];
            rs += __shfl_xor(rs, 1, 32);
            rs += __shfl_xor(rs, 2, 32);
            rs += __shfl_xor(rs, 4, 32);
            rs += __shfl_xor(rs, 8, 32);
            lrow[r] = lrow[r] * alpha[r] + rs;
            o0[r] *= alpha[r]; o1[r] *= alpha[r];
            o2[r] *= alpha[r]; o3[r] *= alpha[r];
        }

        // ---- restage P (C layout -> A layout) through per-wave LDS tile
        _Float16* ps = pst[w];
        #pragma unroll
        for (int r = 0; r < 8; ++r) {
            int row = lg * 8 + r;
            ps[row * 32 + ln]      = (_Float16)s0[r];
            ps[row * 32 + 16 + ln] = (_Float16)s1[r];
        }
        U16 pf;
        {
            const _Float16* pr = ps + ln * 32 + lg * 16;
            pf.h2[0] = *(const v8h*)(pr);
            pf.h2[1] = *(const v8h*)(pr + 8);
        }

        // ---- O += P @ V  (V transposed layout: contiguous along keys)
        {
            v16h vf0 = *(const v16h*)(vb + (size_t)(0 * 16 + ln) * kLK + lg * 16);
            o0 = wmma_f16(pf.v, vf0, o0);
            v16h vf1 = *(const v16h*)(vb + (size_t)(1 * 16 + ln) * kLK + lg * 16);
            o1 = wmma_f16(pf.v, vf1, o1);
            v16h vf2 = *(const v16h*)(vb + (size_t)(2 * 16 + ln) * kLK + lg * 16);
            o2 = wmma_f16(pf.v, vf2, o2);
            v16h vf3 = *(const v16h*)(vb + (size_t)(3 * 16 + ln) * kLK + lg * 16);
            o3 = wmma_f16(pf.v, vf3, o3);
        }
    }

    // ---- normalize and store to (B, LQ, DIM) f16 for the output projection
    #pragma unroll
    for (int r = 0; r < 8; ++r) {
        float inv = 1.0f / lrow[r];
        int q = q0 + lg * 8 + r;
        size_t base = ((size_t)b * kLQ + q) * kDIM + h * kDK;
        Oh[base + 0 * 16 + ln] = (_Float16)(o0[r] * inv);
        Oh[base + 1 * 16 + ln] = (_Float16)(o1[r] * inv);
        Oh[base + 2 * 16 + ln] = (_Float16)(o2[r] * inv);
        Oh[base + 3 * 16 + ln] = (_Float16)(o3[r] * inv);
    }
}

// ---------------------------------------------------------------------------
extern "C" void kernel_launch(void* const* d_in, const int* in_sizes, int n_in,
                              void* d_out, int out_size, void* d_ws, size_t ws_size,
                              hipStream_t stream) {
    const float* query = (const float*)d_in[0];
    const float* key   = (const float*)d_in[1];
    const float* value = (const float*)d_in[2];
    const int*   kpm   = (const int*)d_in[3];
    const int*   am    = (const int*)d_in[4];
    const float* Wq = (const float*)d_in[5];  const float* bq = (const float*)d_in[6];
    const float* Wk = (const float*)d_in[7];  const float* bk = (const float*)d_in[8];
    const float* Wv = (const float*)d_in[9];  const float* bv = (const float*)d_in[10];
    const float* Wo = (const float*)d_in[11]; const float* bo = (const float*)d_in[12];

    char* ws = (char*)d_ws;
    _Float16* Qh = (_Float16*)(ws + (size_t)0);
    _Float16* Kh = (_Float16*)(ws + (size_t)8 * 1024 * 1024);
    _Float16* Vt = (_Float16*)(ws + (size_t)16 * 1024 * 1024);
    _Float16* Oh = (_Float16*)(ws + (size_t)24 * 1024 * 1024);

    dim3 gg(32, 16);   // 4096/128 row tiles x 1024/64 col tiles
    proj_gemm<0, 0><<<gg, 256, 0, stream>>>(query, Wq, bq, Qh);
    proj_gemm<0, 0><<<gg, 256, 0, stream>>>(key,   Wk, bk, Kh);
    proj_gemm<0, 1><<<gg, 256, 0, stream>>>(value, Wv, bv, Vt);

    attn_kernel<<<dim3(kLQ / 64, kB * kHEADS), 128, 0, stream>>>(Qh, Kh, Vt, am, kpm, Oh);

    proj_gemm<1, 2><<<gg, 256, 0, stream>>>(Oh, Wo, bo, d_out);
}
